// Transformer_678604832829
// MI455X (gfx1250) — matrix operand order repaired and checked by reading, not compiled, not run on hardware
//
#include <hip/hip_runtime.h>
#include <stdint.h>

// ---------------------------------------------------------------------------
// Transformer encoder-decoder forward (B=4, L=512, d_model=512, 8 heads x 64,
// d_inner=2048, 6+6 layers), fp32 reference. All matmuls run on CDNA5 WMMA
// (v_wmma_f32_16x16x32_f16): fp16 operands, fp32 accumulation.
//
// GEMM: block tile 128x64, 8 waves, wave tile 32x32 (4 WMMA / 32-K step),
// double-buffered LDS staging via GLOBAL_LOAD_ASYNC_TO_LDS_B128 (ASYNCcnt)
// when the builtin is available, else synchronous staging.
//
// Every GEMM is C[M,N] = A[M,K] . B[N,K]^T ("NT"):
//   - QKV proj: weights [h,d,k] pre-transposed to [h*64+k, d] fp16 (once/call)
//   - scores:   Q[L,64] . K[L,64]^T  (natural NT)
//   - attn.V:   attn[L,L] . Vt[64,L]^T (V transposed by helper kernel)
//   - out proj / FFN: [out,in]-layout weights are already NT
// All M multiples of 128, N multiples of 64, K multiples of 32 -> no guards;
// EXEC is all-ones around every WMMA (ISA requirement).
//
// Input flattening assumption (setup_inputs insertion order, depth first):
//   0 src_seq 1 src_pos 2 tgt_seq 3 tgt_pos 4 src_emb 5 tgt_emb
//   6 src_pos_tab 7 tgt_pos_tab
//   8..14  enc_attn {wq,wk,wv,pw,pb,ln_g,ln_b}
//   15..20 enc_ffn  {w1,b1,w2,b2,ln_g,ln_b}
//   21..27 dec_slf  {wq,wk,wv,pw,pb,ln_g,ln_b}
//   28..34 dec_crs  {wq,wk,wv,pw,pb,ln_g,ln_b}
//   35..40 dec_ffn  {w1,b1,w2,b2,ln_g,ln_b}
// ---------------------------------------------------------------------------

typedef __attribute__((ext_vector_type(16))) _Float16 v16h;
typedef __attribute__((ext_vector_type(8)))  float    v8f;
typedef _Float16 h16;

#define DMODEL   512
#define NHEAD    8
#define DK       64
#define BATCH    4
#define SEQL     512
#define DINNER   2048
#define NLAYERS  6
#define ROWS     (BATCH * SEQL)          // 2048
#define LW       (DMODEL * DMODEL)       // 262144, per-layer attn weight
#define LWF      (DINNER * DMODEL)       // 1048576, per-layer ffn weight
#define EPS      1e-6f
#define INV_TEMPER 0.0441941738241592f   // 1/sqrt(512)

#define TM 128                           // block tile M
#define TN 64                            // block tile N
#define TK 32                            // K step

#if __has_builtin(__builtin_amdgcn_global_load_async_to_lds_b128)
#define USE_ASYNC_LDS 1
#endif

// Builtin expects pointers to <4 x i32> in addrspace(1)/(3).
typedef int v4i __attribute__((vector_size(16)));
typedef __attribute__((address_space(1))) v4i* gptr_t;
typedef __attribute__((address_space(3))) v4i* lptr_t;

__device__ __forceinline__ gptr_t to_global(const void* p) {
    return (gptr_t)(unsigned long long)(uintptr_t)p;   // global generic == as1
}
__device__ __forceinline__ lptr_t to_lds(void* p) {
    return (lptr_t)(unsigned)(uintptr_t)p;             // LDS offset = low 32 bits
}

__device__ __forceinline__ void wait_async0() {
#if __has_builtin(__builtin_amdgcn_s_wait_asynccnt)
    __builtin_amdgcn_s_wait_asynccnt(0);
#else
    asm volatile("s_wait_asynccnt 0" ::: "memory");
#endif
}

// ---------------------------------------------------------------------------
// Generic batched NT WMMA GEMM. FLAGS: 1=bias, 2=relu, 4=scale.
// Batch offset = (bz/bdiv)*s?1 + (bz%bdiv)*s?2  (handles [b][h] batching).
// ---------------------------------------------------------------------------
template<int FLAGS>
__global__ __launch_bounds__(256) void gemm_nt_kernel(
    const h16* __restrict__ A, int lda,
    const h16* __restrict__ B, int ldb,
    float* __restrict__ C, int ldc,
    int K, int bdiv,
    long long sA1, long long sA2, long long sB1, long long sB2,
    long long sC1, long long sC2,
    const float* __restrict__ bias, float scale)
{
    __shared__ __align__(16) h16 sa[2][TM][TK];   // 2 x 8 KB
    __shared__ __align__(16) h16 sb[2][TN][TK];   // 2 x 4 KB

    const int bz = blockIdx.z;
    A += (long long)(bz / bdiv) * sA1 + (long long)(bz % bdiv) * sA2;
    B += (long long)(bz / bdiv) * sB1 + (long long)(bz % bdiv) * sB2;
    C += (long long)(bz / bdiv) * sC1 + (long long)(bz % bdiv) * sC2;

    const int m0 = blockIdx.y * TM;
    const int n0 = blockIdx.x * TN;

    const int tid  = threadIdx.x;
    const int wave = tid >> 5;        // 0..7 (wave32)
    const int lane = tid & 31;
    const int wm   = wave & 3;        // wave M sub-tile (32 rows)
    const int wn   = wave >> 2;       // wave N sub-tile (32 cols)
    const int g    = lane >> 4;       // half-wave group
    const int ln   = lane & 15;

    // Staging decomposition: 16B chunks. A: 128x32 halves = 512 chunks
    // (2 per thread), B: 64x32 = 256 chunks (1 per thread). chunk c ->
    // row = c>>2, col halves = (c&3)*8.
    const int ar0 = tid >> 2,         ac0 = (tid & 3) << 3;
    const int ar1 = (tid + 256) >> 2, ac1 = ((tid + 256) & 3) << 3;
    const int br0 = tid >> 2,         bc0 = (tid & 3) << 3;

    v8f acc00 = {}, acc01 = {}, acc10 = {}, acc11 = {};

    const int nk = K >> 5;

#ifdef USE_ASYNC_LDS
    auto stage = [&](int buf, int k0) {
        __builtin_amdgcn_global_load_async_to_lds_b128(
            to_global(A + (long long)(m0 + ar0) * lda + k0 + ac0),
            to_lds(&sa[buf][ar0][ac0]), 0, 0);
        __builtin_amdgcn_global_load_async_to_lds_b128(
            to_global(A + (long long)(m0 + ar1) * lda + k0 + ac1),
            to_lds(&sa[buf][ar1][ac1]), 0, 0);
        __builtin_amdgcn_global_load_async_to_lds_b128(
            to_global(B + (long long)(n0 + br0) * ldb + k0 + bc0),
            to_lds(&sb[buf][br0][bc0]), 0, 0);
    };
#else
    auto stage = [&](int buf, int k0) {
        *(uint4*)(&sa[buf][ar0][ac0]) = *(const uint4*)(A + (long long)(m0 + ar0) * lda + k0 + ac0);
        *(uint4*)(&sa[buf][ar1][ac1]) = *(const uint4*)(A + (long long)(m0 + ar1) * lda + k0 + ac1);
        *(uint4*)(&sb[buf][br0][bc0]) = *(const uint4*)(B + (long long)(n0 + br0) * ldb + k0 + bc0);
    };
#endif

    auto compute = [&](int buf) {
        union { v16h v; unsigned u[8]; } fa0, fa1, fb0, fb1;
        #pragma unroll
        for (int v = 0; v < 8; ++v) {
            // A 16x32 fp16 fragment: VGPR v holds K = 16*(v>>2) + 8*g + 2*(v&3) .. +1
            const int ka = 16 * (v >> 2) + 8 * g + 2 * (v & 3);
            fa0.u[v] = *(const unsigned*)(&sa[buf][wm * 32 + ln][ka]);
            fa1.u[v] = *(const unsigned*)(&sa[buf][wm * 32 + 16 + ln][ka]);
            // B 32x16 fp16 fragment (column per lane): K = 16*(v>>2) + 8*g + 2*(v&3) .. +1
            const int kb = 16 * (v >> 2) + 8 * g + 2 * (v & 3);
            fb0.u[v] = *(const unsigned*)(&sb[buf][wn * 32 + ln][kb]);
            fb1.u[v] = *(const unsigned*)(&sb[buf][wn * 32 + 16 + ln][kb]);
        }
        acc00 = __builtin_amdgcn_wmma_f32_16x16x32_f16(false, fa0.v, false, fb0.v, (short)0, acc00, false, false);
        acc01 = __builtin_amdgcn_wmma_f32_16x16x32_f16(false, fa0.v, false, fb1.v, (short)0, acc01, false, false);
        acc10 = __builtin_amdgcn_wmma_f32_16x16x32_f16(false, fa1.v, false, fb0.v, (short)0, acc10, false, false);
        acc11 = __builtin_amdgcn_wmma_f32_16x16x32_f16(false, fa1.v, false, fb1.v, (short)0, acc11, false, false);
    };

#ifdef USE_ASYNC_LDS
    // Double-buffered pipeline: prefetch tile i+1 while WMMAs consume tile i.
    stage(0, 0);
    for (int i = 0; i < nk; ++i) {
        const int cur = i & 1;
        wait_async0();        // own async ops done (LDS written)
        __syncthreads();      // whole tile visible; prior readers finished
        if (i + 1 < nk) stage(cur ^ 1, (i + 1) << 5);
        compute(cur);
    }
#else
    for (int i = 0; i < nk; ++i) {
        stage(0, i << 5);
        __syncthreads();
        compute(0);
        __syncthreads();
    }
#endif

    // C/D layout: VGPR r -> row M = r + 8*g, col N = lane%16
    const int colA = n0 + wn * 32 + ln;
    const int colB = colA + 16;
    const int rowA = m0 + wm * 32;
    #pragma unroll
    for (int r = 0; r < 8; ++r) {
        const int r0 = rowA + r + 8 * g;
        const int r1 = r0 + 16;
        float x00 = acc00[r], x01 = acc01[r], x10 = acc10[r], x11 = acc11[r];
        if (FLAGS & 4) { x00 *= scale; x01 *= scale; x10 *= scale; x11 *= scale; }
        if (FLAGS & 1) {
            const float b0 = bias[colA], b1 = bias[colB];
            x00 += b0; x01 += b1; x10 += b0; x11 += b1;
        }
        if (FLAGS & 2) {
            x00 = fmaxf(x00, 0.f); x01 = fmaxf(x01, 0.f);
            x10 = fmaxf(x10, 0.f); x11 = fmaxf(x11, 0.f);
        }
        C[(long long)r0 * ldc + colA] = x00;
        C[(long long)r0 * ldc + colB] = x01;
        C[(long long)r1 * ldc + colA] = x10;
        C[(long long)r1 * ldc + colB] = x11;
    }
}

// ---------------------------------------------------------------------------
// Elementwise / helper kernels
// ---------------------------------------------------------------------------

__global__ void cast_f32_f16_kernel(const float* __restrict__ src,
                                    h16* __restrict__ dst, long long n)
{
    for (long long i = (long long)blockIdx.x * blockDim.x + threadIdx.x; i < n;
         i += (long long)gridDim.x * blockDim.x)
        dst[i] = (h16)src[i];
}

// wq/wk/wv: fp32 [6][8][512][64] (layer,head,d,k) -> fp16 [6][h*64+k][d]
__global__ void conv_wqkv_kernel(const float* __restrict__ src,
                                 h16* __restrict__ dst, long long n)
{
    for (long long i = (long long)blockIdx.x * blockDim.x + threadIdx.x; i < n;
         i += (long long)gridDim.x * blockDim.x) {
        const int d   = (int)(i & 511);
        const int hk  = (int)((i >> 9) & 511);
        const int lay = (int)(i >> 18);
        const int h   = hk >> 6;
        const int k   = hk & 63;
        dst[i] = (h16)src[((((long long)lay * 8 + h) * 512 + d) << 6) + k];
    }
}

// x[b,l,d] = emb[seq[b,l]][d] + pos_tab[pos[b,l]][d]; also fp16 copy
__global__ void embed_kernel(const int* __restrict__ seq, const int* __restrict__ pos,
                             const float* __restrict__ emb, const float* __restrict__ ptab,
                             float* __restrict__ x, h16* __restrict__ xh, long long n)
{
    for (long long i = (long long)blockIdx.x * blockDim.x + threadIdx.x; i < n;
         i += (long long)gridDim.x * blockDim.x) {
        const long long bl = i >> 9;
        const int d = (int)(i & 511);
        const int t = seq[bl];
        const int p = pos[bl];
        const float v = emb[(long long)t * DMODEL + d] + ptab[(long long)p * DMODEL + d];
        x[i]  = v;
        xh[i] = (h16)v;
    }
}

// fp16 [B*L, 512] (col = h*64+v) -> Vt fp16 [B][h][64][L]
__global__ void transpose_v_kernel(const h16* __restrict__ v, h16* __restrict__ vt)
{
    const long long n = (long long)BATCH * NHEAD * DK * SEQL;
    for (long long i = (long long)blockIdx.x * blockDim.x + threadIdx.x; i < n;
         i += (long long)gridDim.x * blockDim.x) {
        const int l = (int)(i & 511);
        const int k = (int)((i >> 9) & 63);
        const int h = (int)((i >> 15) & 7);
        const int b = (int)(i >> 18);
        vt[i] = v[((long long)(b * SEQL + l) * DMODEL) + h * DK + k];
    }
}

// Masked softmax over rows of scores [B][h][Lq][Lk=512]; one block per row.
// mask: key pad (kseq[b,k]==0) and optional causal (k > q). Writes fp16 attn.
__global__ void softmax_kernel(const float* __restrict__ scores,
                               const int* __restrict__ kseq,
                               h16* __restrict__ attn, int causal)
{
    const int r = blockIdx.x;                 // [0, B*8*512)
    const int b = r >> 12;
    const int q = r & 511;
    const float* sr = scores + (long long)r * SEQL;
    h16* ar = attn + (long long)r * SEQL;
    const int t = threadIdx.x;                // 256 threads, 2 elems each

    float v0 = sr[t];
    float v1 = sr[t + 256];
    const bool m0 = (kseq[b * SEQL + t] == 0)       || (causal && (t > q));
    const bool m1 = (kseq[b * SEQL + t + 256] == 0) || (causal && (t + 256 > q));
    if (m0) v0 = -1e9f;
    if (m1) v1 = -1e9f;

    __shared__ float red[256];
    red[t] = fmaxf(v0, v1);
    __syncthreads();
    for (int s = 128; s > 0; s >>= 1) {
        if (t < s) red[t] = fmaxf(red[t], red[t + s]);
        __syncthreads();
    }
    const float mx = red[0];
    __syncthreads();

    const float e0 = __expf(v0 - mx);
    const float e1 = __expf(v1 - mx);
    red[t] = e0 + e1;
    __syncthreads();
    for (int s = 128; s > 0; s >>= 1) {
        if (t < s) red[t] += red[t + s];
        __syncthreads();
    }
    const float inv = 1.0f / red[0];
    ar[t]       = (h16)(e0 * inv);
    ar[t + 256] = (h16)(e1 * inv);
}

// y = LN(gin + bias + x) * gamma + beta ; writes fp32 back into x and fp16 xh.
// One block (256 thr) per row, D = 512.
__global__ void bias_resid_ln_kernel(const float* __restrict__ gin,
                                     const float* __restrict__ bias,
                                     float* __restrict__ x,
                                     const float* __restrict__ gamma,
                                     const float* __restrict__ beta,
                                     h16* __restrict__ xh)
{
    const int row = blockIdx.x;
    const int t   = threadIdx.x;
    const float* gr = gin + (long long)row * DMODEL;
    float* xr = x + (long long)row * DMODEL;

    float v0 = gr[t]       + bias[t]       + xr[t];
    float v1 = gr[t + 256] + bias[t + 256] + xr[t + 256];

    __shared__ float red[256];
    red[t] = v0 + v1;
    __syncthreads();
    for (int s = 128; s > 0; s >>= 1) {
        if (t < s) red[t] += red[t + s];
        __syncthreads();
    }
    const float mean = red[0] * (1.0f / DMODEL);
    __syncthreads();

    const float d0 = v0 - mean, d1 = v1 - mean;
    red[t] = d0 * d0 + d1 * d1;
    __syncthreads();
    for (int s = 128; s > 0; s >>= 1) {
        if (t < s) red[t] += red[t + s];
        __syncthreads();
    }
    const float inv = rsqrtf(red[0] * (1.0f / DMODEL) + EPS);

    const float o0 = gamma[t] * d0 * inv + beta[t];
    const float o1 = gamma[t + 256] * d1 * inv + beta[t + 256];
    xr[t]       = o0;
    xr[t + 256] = o1;
    xh[(long long)row * DMODEL + t]       = (h16)o0;
    xh[(long long)row * DMODEL + t + 256] = (h16)o1;
}

// ---------------------------------------------------------------------------
// Host-side orchestration
// ---------------------------------------------------------------------------

namespace {

struct AttnW {
    const h16 *wq, *wk, *wv, *pw;               // fp16, per-layer strided
    const float *pb, *lng, *lnb;                // fp32 raw inputs
};
struct FfnW {
    const h16 *w1, *w2;
    const float *b1, *b2, *lng, *lnb;
};

struct Ws {
    // fp16 converted weights
    h16 *ea_wq, *ea_wk, *ea_wv, *ea_pw;
    h16 *ds_wq, *ds_wk, *ds_wv, *ds_pw;
    h16 *dc_wq, *dc_wk, *dc_wv, *dc_pw;
    h16 *ef_w1, *ef_w2, *df_w1, *df_w2;
    // activations
    float *enc_x, *dec_x, *gout, *scores;
    h16 *enc_xh, *dec_xh, *qh, *kh, *vh, *vt, *attnh, *cath, *hidh;
};

inline int gblocks(long long n) {
    long long b = (n + 255) / 256;
    return (int)(b > 1048576 ? 1048576 : b);
}

template<int FLAGS>
void launch_gemm(const h16* A, int lda, const h16* B, int ldb, float* C, int ldc,
                 int M, int N, int K, int batch, int bdiv,
                 long long sA1, long long sA2, long long sB1, long long sB2,
                 long long sC1, long long sC2,
                 const float* bias, float scale, hipStream_t s)
{
    dim3 grid(N / TN, M / TM, batch);
    gemm_nt_kernel<FLAGS><<<grid, 256, 0, s>>>(A, lda, B, ldb, C, ldc, K, bdiv,
                                               sA1, sA2, sB1, sB2, sC1, sC2,
                                               bias, scale);
}

void cast_f(const float* src, h16* dst, long long n, hipStream_t s) {
    cast_f32_f16_kernel<<<gblocks(n), 256, 0, s>>>(src, dst, n);
}

// Multi-head attention block: query/residual = (x, xh); key/value source = kvh.
void run_mha(float* x, h16* xh, const h16* kvh, const AttnW& W, int lay,
             const int* key_seq, int causal, Ws& ws, hipStream_t s)
{
    // Q/K/V projections: [2048,512] x [512,512]^T
    launch_gemm<0>(xh, DMODEL, W.wq + (long long)lay * LW, DMODEL, ws.gout, DMODEL,
                   ROWS, DMODEL, DMODEL, 1, 1, 0, 0, 0, 0, 0, 0, nullptr, 0.f, s);
    cast_f(ws.gout, ws.qh, (long long)ROWS * DMODEL, s);
    launch_gemm<0>(kvh, DMODEL, W.wk + (long long)lay * LW, DMODEL, ws.gout, DMODEL,
                   ROWS, DMODEL, DMODEL, 1, 1, 0, 0, 0, 0, 0, 0, nullptr, 0.f, s);
    cast_f(ws.gout, ws.kh, (long long)ROWS * DMODEL, s);
    launch_gemm<0>(kvh, DMODEL, W.wv + (long long)lay * LW, DMODEL, ws.gout, DMODEL,
                   ROWS, DMODEL, DMODEL, 1, 1, 0, 0, 0, 0, 0, 0, nullptr, 0.f, s);
    cast_f(ws.gout, ws.vh, (long long)ROWS * DMODEL, s);
    transpose_v_kernel<<<4096, 256, 0, s>>>(ws.vh, ws.vt);

    // scores[b][h] = Q[b,h] . K[b,h]^T / temper : batch = 32 (b major, h minor)
    launch_gemm<4>(ws.qh, DMODEL, ws.kh, DMODEL, ws.scores, SEQL,
                   SEQL, SEQL, DK, BATCH * NHEAD, NHEAD,
                   (long long)SEQL * DMODEL, DK,                 // A: per-b, per-h
                   (long long)SEQL * DMODEL, DK,                 // B: per-b, per-h
                   (long long)NHEAD * SEQL * SEQL, (long long)SEQL * SEQL,
                   nullptr, INV_TEMPER, s);

    softmax_kernel<<<BATCH * NHEAD * SEQL, 256, 0, s>>>(ws.scores, key_seq, ws.attnh, causal);

    // concat[b,q][h*64+v] = attn[b,h] . Vt[b,h]^T
    launch_gemm<0>(ws.attnh, SEQL, ws.vt, SEQL, ws.gout, DMODEL,
                   SEQL, DK, SEQL, BATCH * NHEAD, NHEAD,
                   (long long)NHEAD * SEQL * SEQL, (long long)SEQL * SEQL,
                   (long long)NHEAD * DK * SEQL, (long long)DK * SEQL,
                   (long long)SEQL * DMODEL, DK,
                   nullptr, 0.f, s);
    cast_f(ws.gout, ws.cath, (long long)ROWS * DMODEL, s);

    // output projection (pw is [d_out, j] -> already NT)
    launch_gemm<0>(ws.cath, DMODEL, W.pw + (long long)lay * LW, DMODEL, ws.gout, DMODEL,
                   ROWS, DMODEL, DMODEL, 1, 1, 0, 0, 0, 0, 0, 0, nullptr, 0.f, s);

    bias_resid_ln_kernel<<<ROWS, 256, 0, s>>>(ws.gout, W.pb + lay * DMODEL, x,
                                              W.lng + lay * DMODEL, W.lnb + lay * DMODEL, xh);
}

void run_ffn(float* x, h16* xh, const FfnW& W, int lay, Ws& ws, hipStream_t s)
{
    // h = relu(x . w1^T + b1) : [2048,2048]
    launch_gemm<3>(xh, DMODEL, W.w1 + (long long)lay * LWF, DMODEL, ws.gout, DINNER,
                   ROWS, DINNER, DMODEL, 1, 1, 0, 0, 0, 0, 0, 0,
                   W.b1 + lay * DINNER, 0.f, s);
    cast_f(ws.gout, ws.hidh, (long long)ROWS * DINNER, s);
    // out = h . w2^T : [2048,512]
    launch_gemm<0>(ws.hidh, DINNER, W.w2 + (long long)lay * LWF, DINNER, ws.gout, DMODEL,
                   ROWS, DMODEL, DINNER, 1, 1, 0, 0, 0, 0, 0, 0, nullptr, 0.f, s);
    bias_resid_ln_kernel<<<ROWS, 256, 0, s>>>(ws.gout, W.b2 + lay * DMODEL, x,
                                              W.lng + lay * DMODEL, W.lnb + lay * DMODEL, xh);
}

} // namespace

extern "C" void kernel_launch(void* const* d_in, const int* in_sizes, int n_in,
                              void* d_out, int out_size, void* d_ws, size_t ws_size,
                              hipStream_t stream)
{
    (void)in_sizes; (void)n_in; (void)out_size; (void)ws_size;

    const int*   src_seq  = (const int*)d_in[0];
    const int*   src_pos  = (const int*)d_in[1];
    const int*   tgt_seq  = (const int*)d_in[2];
    const int*   tgt_pos  = (const int*)d_in[3];
    const float* src_emb  = (const float*)d_in[4];
    const float* tgt_emb  = (const float*)d_in[5];
    const float* src_ptab = (const float*)d_in[6];
    const float* tgt_ptab = (const float*)d_in[7];

    // params, flattened depth-first in insertion order
    const float* ea[7]; const float* ef[6]; const float* dsl[7];
    const float* dcr[7]; const float* df[6];
    for (int i = 0; i < 7; ++i) ea[i]  = (const float*)d_in[8 + i];
    for (int i = 0; i < 6; ++i) ef[i]  = (const float*)d_in[15 + i];
    for (int i = 0; i < 7; ++i) dsl[i] = (const float*)d_in[21 + i];
    for (int i = 0; i < 7; ++i) dcr[i] = (const float*)d_in[28 + i];
    for (int i = 0; i < 6; ++i) df[i]  = (const float*)d_in[35 + i];

    // ---- workspace bump allocator -----------------------------------------
    char* base = (char*)d_ws;
    size_t off = 0;
    auto alloc = [&](size_t bytes) -> void* {
        void* p = base + off;
        off += (bytes + 255) & ~(size_t)255;
        return p;
    };

    const size_t AW = (size_t)NLAYERS * LW;       // 1,572,864 elems (attn weight)
    const size_t FW = (size_t)NLAYERS * LWF;      // 6,291,456 elems (ffn weight)

    Ws ws;
    ws.ea_wq = (h16*)alloc(AW * 2); ws.ea_wk = (h16*)alloc(AW * 2);
    ws.ea_wv = (h16*)alloc(AW * 2); ws.ea_pw = (h16*)alloc(AW * 2);
    ws.ds_wq = (h16*)alloc(AW * 2); ws.ds_wk = (h16*)alloc(AW * 2);
    ws.ds_wv = (h16*)alloc(AW * 2); ws.ds_pw = (h16*)alloc(AW * 2);
    ws.dc_wq = (h16*)alloc(AW * 2); ws.dc_wk = (h16*)alloc(AW * 2);
    ws.dc_wv = (h16*)alloc(AW * 2); ws.dc_pw = (h16*)alloc(AW * 2);
    ws.ef_w1 = (h16*)alloc(FW * 2); ws.ef_w2 = (h16*)alloc(FW * 2);
    ws.df_w1 = (h16*)alloc(FW * 2); ws.df_w2 = (h16*)alloc(FW * 2);

    const size_t NX = (size_t)ROWS * DMODEL;      // 1,048,576
    const size_t NS = (size_t)BATCH * NHEAD * SEQL * SEQL; // 8,388,608
    const size_t NH = (size_t)ROWS * DINNER;      // 4,194,304

    ws.enc_x  = (float*)alloc(NX * 4);
    ws.dec_x  = (float*)alloc(NX * 4);
    ws.gout   = (float*)alloc(NH * 4);            // max GEMM fp32 output
    ws.scores = (float*)alloc(NS * 4);
    ws.enc_xh = (h16*)alloc(NX * 2);
    ws.dec_xh = (h16*)alloc(NX * 2);
    ws.qh     = (h16*)alloc(NX * 2);
    ws.kh     = (h16*)alloc(NX * 2);
    ws.vh     = (h16*)alloc(NX * 2);
    ws.vt     = (h16*)alloc(NX * 2);
    ws.cath   = (h16*)alloc(NX * 2);
    ws.attnh  = (h16*)alloc(NS * 2);
    ws.hidh   = (h16*)alloc(NH * 2);

    // ---- weight conversion (fp32 -> fp16, QKV transposed to [h*64+k, d]) ---
    conv_wqkv_kernel<<<gblocks(AW), 256, 0, stream>>>(ea[0],  ws.ea_wq, (long long)AW);
    conv_wqkv_kernel<<<gblocks(AW), 256, 0, stream>>>(ea[1],  ws.ea_wk, (long long)AW);
    conv_wqkv_kernel<<<gblocks(AW), 256, 0, stream>>>(ea[2],  ws.ea_wv, (long long)AW);
    cast_f(ea[3], ws.ea_pw, (long long)AW, stream);
    conv_wqkv_kernel<<<gblocks(AW), 256, 0, stream>>>(dsl[0], ws.ds_wq, (long long)AW);
    conv_wqkv_kernel<<<gblocks(AW), 256, 0, stream>>>(dsl[1], ws.ds_wk, (long long)AW);
    conv_wqkv_kernel<<<gblocks(AW), 256, 0, stream>>>(dsl[2], ws.ds_wv, (long long)AW);
    cast_f(dsl[3], ws.ds_pw, (long long)AW, stream);
    conv_wqkv_kernel<<<gblocks(AW), 256, 0, stream>>>(dcr[0], ws.dc_wq, (long long)AW);
    conv_wqkv_kernel<<<gblocks(AW), 256, 0, stream>>>(dcr[1], ws.dc_wk, (long long)AW);
    conv_wqkv_kernel<<<gblocks(AW), 256, 0, stream>>>(dcr[2], ws.dc_wv, (long long)AW);
    cast_f(dcr[3], ws.dc_pw, (long long)AW, stream);
    cast_f(ef[0], ws.ef_w1, (long long)FW, stream);
    cast_f(ef[2], ws.ef_w2, (long long)FW, stream);
    cast_f(df[0], ws.df_w1, (long long)FW, stream);
    cast_f(df[2], ws.df_w2, (long long)FW, stream);

    AttnW enc_attn = { ws.ea_wq, ws.ea_wk, ws.ea_wv, ws.ea_pw, ea[4],  ea[5],  ea[6]  };
    AttnW dec_slf  = { ws.ds_wq, ws.ds_wk, ws.ds_wv, ws.ds_pw, dsl[4], dsl[5], dsl[6] };
    AttnW dec_crs  = { ws.dc_wq, ws.dc_wk, ws.dc_wv, ws.dc_pw, dcr[4], dcr[5], dcr[6] };
    FfnW  enc_ffn  = { ws.ef_w1, ws.ef_w2, ef[1], ef[3], ef[4], ef[5] };
    FfnW  dec_ffn  = { ws.df_w1, ws.df_w2, df[1], df[3], df[4], df[5] };

    // ---- encoder ----------------------------------------------------------
    embed_kernel<<<4096, 256, 0, stream>>>(src_seq, src_pos, src_emb, src_ptab,
                                           ws.enc_x, ws.enc_xh, (long long)NX);
    for (int i = 0; i < NLAYERS; ++i) {
        run_mha(ws.enc_x, ws.enc_xh, ws.enc_xh, enc_attn, i, src_seq, 0, ws, stream);
        run_ffn(ws.enc_x, ws.enc_xh, enc_ffn, i, ws, stream);
    }

    // ---- decoder ----------------------------------------------------------
    embed_kernel<<<4096, 256, 0, stream>>>(tgt_seq, tgt_pos, tgt_emb, tgt_ptab,
                                           ws.dec_x, ws.dec_xh, (long long)NX);
    for (int i = 0; i < NLAYERS; ++i) {
        run_mha(ws.dec_x, ws.dec_xh, ws.dec_xh, dec_slf, i, tgt_seq, 1, ws, stream);
        run_mha(ws.dec_x, ws.dec_xh, ws.enc_xh, dec_crs, i, src_seq, 0, ws, stream);
        run_ffn(ws.dec_x, ws.dec_xh, dec_ffn, i, ws, stream);
    }

    (void)hipMemcpyAsync(d_out, ws.dec_x, NX * sizeof(float),
                         hipMemcpyDeviceToDevice, stream);
}
